// CosineSimLayer_12103217840496
// MI455X (gfx1250) — compile-verified, matrix operand order checked by reading, not run on hardware
//
#include <hip/hip_runtime.h>
#include <hip/hip_bf16.h>
#include <stddef.h>
#include <stdint.h>

// ---------------------------------------------------------------------------
// Types for CDNA5 WMMA (wave32): 16x16x32 bf16, f32 accumulate
// ---------------------------------------------------------------------------
typedef __bf16 bf16;
typedef __bf16 bfx16 __attribute__((ext_vector_type(16)));
typedef __bf16 bfx8  __attribute__((ext_vector_type(8)));
typedef float  fx8   __attribute__((ext_vector_type(8)));

union FragAB { bfx16 v; bfx8 h[2]; };

#define GEMM_TILE   128
#define LDS_K       32
#define LDS_STRIDE  40   // 32 + 8 bf16 pad -> 80B row stride (16B aligned, conflict-spread)
#define STAGE_ELEMS (GEMM_TILE * LDS_STRIDE)
#define NBUF        3    // triple buffer -> single barrier per K-step

// CDNA5 async memory->LDS DMA (tracked by ASYNCcnt).
// VDST operand = per-lane LDS byte offset, VADDR = 64-bit global address.
__device__ __forceinline__ void async_load_b128(uint32_t lds_off, const void* gaddr)
{
    asm volatile("global_load_async_to_lds_b128 %0, %1, off"
                 :: "v"(lds_off), "v"(gaddr) : "memory");
}
__device__ __forceinline__ void wait_async_le4()
{
    asm volatile("s_wait_asynccnt 0x4" ::: "memory");
}

// ---------------------------------------------------------------------------
// Reusable bf16 GEMM:  C[M,N] = A[M,K] * Bm[N,K]^T   (both row-major, A.Bt form)
// Block 128x128, 256 threads = 8 waves in a 2x4 grid; each wave computes a
// 64x32 region = 4x2 tiles of 16x16 via v_wmma_f32_16x16x32_bf16.
// Staging: triple-buffered global_load_async_to_lds_b128, branch-free ring,
// one barrier per K-step. No load guards: M and the B-buffer row count must be
// multiples of 128; K a multiple of 32 with K >= 64. N = store/bias guard only.
// EPI: 0 = store f32, 1 = store bf16, 2 = relu(x+bias)->bf16, 3 = (x+bias)->f32
// ---------------------------------------------------------------------------
template <int EPI>
__global__ __launch_bounds__(256)
void gemm_bf16_abt(const bf16* __restrict__ A, const bf16* __restrict__ Bm,
                   const float* __restrict__ bias, void* __restrict__ Cp,
                   int M, int N, int K, int ldc)
{
    __shared__ __align__(16) bf16 lA[NBUF * STAGE_ELEMS];
    __shared__ __align__(16) bf16 lB[NBUF * STAGE_ELEMS];

    const int tid  = threadIdx.x;
    const int lane = tid & 31;
    const int wave = tid >> 5;        // 0..7
    const int wr   = wave >> 2;       // 0..1 : which 64-row half
    const int wc   = wave & 3;        // 0..3 : which 32-col quarter
    const int ln   = lane & 15;
    const int kb   = (lane >> 4) * 8; // element offset in K for this half-wave
    const int hs   = (lane >> 4) * 8; // C row offset for lanes 16..31

    const int m0 = blockIdx.y * GEMM_TILE;
    const int n0 = blockIdx.x * GEMM_TILE;

    // Per-thread staging chunk geometry: 512 16B-chunks per matrix, 2 per thread.
    const int c0r = tid >> 2,         c0c = (tid & 3) << 3;          // chunk tid
    const int c1r = (tid + 256) >> 2, c1c = ((tid + 256) & 3) << 3;  // chunk tid+256

    // LDS byte offsets (low 32 bits of the flat shared address = LDS offset).
    const uint32_t lA_base = (uint32_t)(uintptr_t)(&lA[0]);
    const uint32_t lB_base = (uint32_t)(uintptr_t)(&lB[0]);
    const uint32_t offA0 = lA_base + (uint32_t)(c0r * LDS_STRIDE + c0c) * 2u;
    const uint32_t offA1 = lA_base + (uint32_t)(c1r * LDS_STRIDE + c1c) * 2u;
    const uint32_t offB0 = lB_base + (uint32_t)(c0r * LDS_STRIDE + c0c) * 2u;
    const uint32_t offB1 = lB_base + (uint32_t)(c1r * LDS_STRIDE + c1c) * 2u;

    const bf16* gA0 = A  + (size_t)(m0 + c0r) * K + c0c;
    const bf16* gA1 = A  + (size_t)(m0 + c1r) * K + c1c;
    const bf16* gB0 = Bm + (size_t)(n0 + c0r) * K + c0c;
    const bf16* gB1 = Bm + (size_t)(n0 + c1r) * K + c1c;

    const uint32_t stageB = (uint32_t)STAGE_ELEMS * 2u;  // bytes per stage

    auto issue_stage = [&](int k0, int buf) {
        const uint32_t sb = (uint32_t)buf * stageB;
        async_load_b128(offA0 + sb, gA0 + k0);
        async_load_b128(offB0 + sb, gB0 + k0);
        async_load_b128(offA1 + sb, gA1 + k0);
        async_load_b128(offB1 + sb, gB1 + k0);
    };

    fx8 acc[4][2];
#pragma unroll
    for (int a = 0; a < 4; ++a)
#pragma unroll
        for (int b = 0; b < 2; ++b)
#pragma unroll
            for (int e = 0; e < 8; ++e) acc[a][b][e] = 0.0f;

    const int nsteps = K / LDS_K;      // >= 2 for all call sites
    issue_stage(0, 0);

    int b_cur = 0;
    for (int s = 0; s < nsteps; ++s) {
        // Branch-free prefetch of stage s+1 (clamped junk re-fetch on the last
        // iteration; it targets a buffer nobody reads and drains at endpgm).
        int snext = s + 1; if (snext >= nsteps) snext = nsteps - 1;
        int b_next = b_cur + 1; if (b_next == NBUF) b_next = 0;
        issue_stage(snext * LDS_K, b_next);

        wait_async_le4();              // retire this wave's stage-s loads
        __syncthreads();               // stage s visible to all waves
        // Safety of single barrier: buffer b_next was last read at step s-2;
        // the step s-1 barrier (already passed) guarantees all waves finished it.

        const bf16* bufA = lA + b_cur * STAGE_ELEMS;
        const bf16* bufB = lB + b_cur * STAGE_ELEMS;
        FragAB fa[4], fb[2];
#pragma unroll
        for (int tr = 0; tr < 4; ++tr) {
            const bf16* p = bufA + (wr * 64 + tr * 16 + ln) * LDS_STRIDE + kb;
            fa[tr].h[0] = *(const bfx8*)(p);
            fa[tr].h[1] = *(const bfx8*)(p + 16);
        }
#pragma unroll
        for (int tc = 0; tc < 2; ++tc) {
            const bf16* p = bufB + (wc * 32 + tc * 16 + ln) * LDS_STRIDE + kb;
            fb[tc].h[0] = *(const bfx8*)(p);
            fb[tc].h[1] = *(const bfx8*)(p + 16);
        }
#pragma unroll
        for (int tr = 0; tr < 4; ++tr)
#pragma unroll
            for (int tc = 0; tc < 2; ++tc)
                acc[tr][tc] = __builtin_amdgcn_wmma_f32_16x16x32_bf16(
                    false, fa[tr].v, false, fb[tc].v,
                    (short)0, acc[tr][tc], false, false);

        b_cur = b_next;
    }

    // ---- epilogue: C/D layout = VGPR v -> row v + (lane<16?0:8), col = lane&15
#pragma unroll
    for (int tr = 0; tr < 4; ++tr) {
#pragma unroll
        for (int tc = 0; tc < 2; ++tc) {
            const int mbase = m0 + wr * 64 + tr * 16 + hs;
            const int nn    = n0 + wc * 32 + tc * 16 + ln;
            if (nn >= N) continue;
            float bv = 0.0f;
            if (EPI >= 2) bv = bias[nn];
#pragma unroll
            for (int v = 0; v < 8; ++v) {
                const int row = mbase + v;
                float x = acc[tr][tc][v];
                if (EPI == 0) {
                    ((float*)Cp)[(size_t)row * ldc + nn] = x;
                } else if (EPI == 1) {
                    ((bf16*)Cp)[(size_t)row * ldc + nn] = (bf16)x;
                } else if (EPI == 2) {
                    x += bv; x = x > 0.0f ? x : 0.0f;
                    ((bf16*)Cp)[(size_t)row * ldc + nn] = (bf16)x;
                } else {
                    ((float*)Cp)[(size_t)row * ldc + nn] = x + bv;
                }
            }
        }
    }
}

// ---------------------------------------------------------------------------
// Row L2-normalize (f32 in) -> bf16 normalized out; optionally also cast the
// raw row into out_raw (used to build final = [enc | memvec] without a concat)
// ---------------------------------------------------------------------------
__global__ __launch_bounds__(256)
void rownorm_cast(const float* __restrict__ in, bf16* __restrict__ out_norm,
                  bf16* __restrict__ out_raw, int D, int raw_ld)
{
    const int row = blockIdx.x;
    const int tid = threadIdx.x;
    float vals[8];
    float acc = 0.0f;
    int c = 0;
    for (int d = tid; d < D; d += 256, ++c) {
        float v = in[(size_t)row * D + d];
        vals[c] = v;
        acc += v * v;
    }
    __shared__ float red[256];
    red[tid] = acc;
    __syncthreads();
    for (int s = 128; s > 0; s >>= 1) {
        if (tid < s) red[tid] += red[tid + s];
        __syncthreads();
    }
    const float rs = 1.0f / sqrtf(red[0] + 1e-6f);
    c = 0;
    for (int d = tid; d < D; d += 256, ++c) {
        out_norm[(size_t)row * D + d] = (bf16)(vals[c] * rs);
        if (out_raw) out_raw[(size_t)row * raw_ld + d] = (bf16)vals[c];
    }
}

// ---------------------------------------------------------------------------
// Tiled transpose + cast: in[R,C] f32 -> out[C,R] bf16
// ---------------------------------------------------------------------------
__global__ __launch_bounds__(256)
void transpose_cast(const float* __restrict__ in, bf16* __restrict__ out,
                    int R, int C)
{
    __shared__ float tile[32][33];
    const int tx = threadIdx.x;           // 0..31
    const int ty = threadIdx.y;           // 0..7
    const int c0 = blockIdx.x * 32;
    const int r0 = blockIdx.y * 32;
#pragma unroll
    for (int i = 0; i < 4; ++i) {
        const int r = r0 + ty + 8 * i;
        tile[ty + 8 * i][tx] = in[(size_t)r * C + c0 + tx];
    }
    __syncthreads();
#pragma unroll
    for (int i = 0; i < 4; ++i) {
        const int cc = c0 + ty + 8 * i;
        out[(size_t)cc * R + r0 + tx] = (bf16)tile[tx][ty + 8 * i];
    }
}

// ---------------------------------------------------------------------------
// Elementwise cast f32 -> bf16
// ---------------------------------------------------------------------------
__global__ __launch_bounds__(256)
void cast_f32_bf16(const float* __restrict__ in, bf16* __restrict__ out, size_t n)
{
    size_t i = (size_t)blockIdx.x * blockDim.x + threadIdx.x;
    size_t stride = (size_t)gridDim.x * blockDim.x;
    for (; i < n; i += stride) out[i] = (bf16)in[i];
}

// ---------------------------------------------------------------------------
// Sparsemax over rows of length 8192 (one 512-thread workgroup per row).
// sparsemax is shift-invariant, so we feed raw cosine scores directly.
// Bitonic sort (descending) in LDS, in-place inclusive scan, k via atomicMax,
// then w = max(z - tau, 0) -> bf16.
// ---------------------------------------------------------------------------
#define SM_N 8192
#define SM_T 512

__global__ __launch_bounds__(SM_T)
void sparsemax_row(const float* __restrict__ S, bf16* __restrict__ W)
{
    __shared__ float s[SM_N];
    __shared__ float part[SM_T];
    __shared__ int   kmax;
    __shared__ float tau_s;

    const int row = blockIdx.x;
    const int tid = threadIdx.x;
    const size_t base = (size_t)row * SM_N;

    if (tid == 0) kmax = 0;
    for (int i = tid; i < SM_N; i += SM_T) s[i] = S[base + i];
    __syncthreads();

    // bitonic sort, descending
    for (int k = 2; k <= SM_N; k <<= 1) {
        for (int j = k >> 1; j > 0; j >>= 1) {
            for (int t = tid; t < SM_N / 2; t += SM_T) {
                const int i = 2 * t - (t & (j - 1));
                const int p = i + j;
                const float a = s[i], b = s[p];
                const bool up = ((i & k) == 0);
                if (up ? (a < b) : (a > b)) { s[i] = b; s[p] = a; }
            }
            __syncthreads();
        }
    }

    // in-place inclusive scan: per-thread serial over 16 elems + scan of partials
    {
        const int seg = tid * (SM_N / SM_T);
        float run = 0.0f;
#pragma unroll
        for (int q = 0; q < SM_N / SM_T; ++q) {
            run += s[seg + q];
            s[seg + q] = run;
        }
        part[tid] = run;
        __syncthreads();
        for (int off = 1; off < SM_T; off <<= 1) {
            const float mine = part[tid];
            const float add  = (tid >= off) ? part[tid - off] : 0.0f;
            __syncthreads();
            part[tid] = mine + add;
            __syncthreads();
        }
        const float offs = (tid > 0) ? part[tid - 1] : 0.0f;
#pragma unroll
        for (int q = 0; q < SM_N / SM_T; ++q) s[seg + q] += offs;
        __syncthreads();
    }

    // support size: z_sorted[i] = s[i]-s[i-1]; support cond: 1 + r*z > cssv
    {
        const int seg = tid * (SM_N / SM_T);
        int local_k = 0;
#pragma unroll
        for (int q = 0; q < SM_N / SM_T; ++q) {
            const int i = seg + q;
            const float cs   = s[i];
            const float prev = (i > 0) ? s[i - 1] : 0.0f;
            const float zi   = cs - prev;
            const float r    = (float)(i + 1);
            if (1.0f + r * zi > cs) local_k = i + 1;
        }
        atomicMax(&kmax, local_k);
        __syncthreads();
        if (tid == 0) {
            const int kk = kmax;
            tau_s = (s[kk - 1] - 1.0f) / (float)kk;
        }
        __syncthreads();
    }

    const float tau = tau_s;
    for (int i = tid; i < SM_N; i += SM_T) {
        const float w = S[base + i] - tau;        // re-read original z (L2-hot)
        W[base + i] = (bf16)(w > 0.0f ? w : 0.0f);
    }
}

// ---------------------------------------------------------------------------
// Launch: B=4096, M=8192, D=1024, OUT=1000
// ---------------------------------------------------------------------------
extern "C" void kernel_launch(void* const* d_in, const int* in_sizes, int n_in,
                              void* d_out, int out_size, void* d_ws, size_t ws_size,
                              hipStream_t stream)
{
    constexpr int Bn = 4096, Mm = 8192, Dd = 1024, ON = 1000;
    constexpr int H1 = 4 * Dd;     // 4096
    constexpr int K1 = 2 * Dd;     // 2048
    constexpr int ONp = 1024;      // fc2 weight rows padded to multiple of 128

    const float* enc  = (const float*)d_in[0];
    const float* mem  = (const float*)d_in[1];
    const float* fc1w = (const float*)d_in[2];
    const float* fc1b = (const float*)d_in[3];
    const float* fc2w = (const float*)d_in[4];
    const float* fc2b = (const float*)d_in[5];
    float* out = (float*)d_out;

    char* ws = (char*)d_ws;
    size_t off = 0;
    auto carve = [&](size_t bytes) -> void* {
        void* p = ws + off;
        off = (off + bytes + 255) & ~(size_t)255;
        return p;
    };

    bf16*  xnorm  = (bf16*)carve((size_t)Bn * Dd * 2);    //  8 MB
    bf16*  ynorm  = (bf16*)carve((size_t)Mm * Dd * 2);    // 16 MB
    bf16*  memT   = (bf16*)carve((size_t)Dd * Mm * 2);    // 16 MB
    float* Smat   = (float*)carve((size_t)Bn * Mm * 4);   // 128 MB
    bf16*  Wmat   = (bf16*)carve((size_t)Bn * Mm * 2);    // 64 MB
    bf16*  finalb = (bf16*)carve((size_t)Bn * K1 * 2);    // 16 MB
    bf16*  fc1wb  = (bf16*)carve((size_t)H1 * K1 * 2);    // 16 MB
    bf16*  hidden = (bf16*)carve((size_t)Bn * H1 * 2);    // 32 MB
    bf16*  fc2wb  = (bf16*)carve((size_t)ONp * H1 * 2);   //  8 MB (padded rows)

    // 1) normalize enc -> xnorm (bf16), raw enc -> final[:, 0:D]
    rownorm_cast<<<Bn, 256, 0, stream>>>(enc, xnorm, finalb, Dd, K1);
    // 2) normalize mem -> ynorm
    rownorm_cast<<<Mm, 256, 0, stream>>>(mem, ynorm, (bf16*)nullptr, Dd, 0);
    // 3) mem^T (bf16) for GEMM2's A.Bt form
    transpose_cast<<<dim3(Dd / 32, Mm / 32), dim3(32, 8), 0, stream>>>(mem, memT, Mm, Dd);
    // 4) weight casts
    cast_f32_bf16<<<1024, 256, 0, stream>>>(fc1w, fc1wb, (size_t)H1 * K1);
    cast_f32_bf16<<<1024, 256, 0, stream>>>(fc2w, fc2wb, (size_t)ON * H1);

    // 5) cos = xnorm @ ynorm^T  -> S [B, M] f32  (sparsemax is shift-invariant)
    gemm_bf16_abt<0><<<dim3(Mm / GEMM_TILE, Bn / GEMM_TILE), 256, 0, stream>>>(
        xnorm, ynorm, nullptr, Smat, Bn, Mm, Dd, Mm);

    // 6) sparsemax per row -> W [B, M] bf16
    sparsemax_row<<<Bn, SM_T, 0, stream>>>(Smat, Wmat);

    // 7) memory_vector = W @ mem = W @ memT^T -> final[:, D:2D] bf16
    gemm_bf16_abt<1><<<dim3(Dd / GEMM_TILE, Bn / GEMM_TILE), 256, 0, stream>>>(
        Wmat, memT, nullptr, finalb + Dd, Bn, Dd, Mm, K1);

    // 8) hidden = relu(final @ fc1_w^T + b1) -> bf16 [B, 4096]
    gemm_bf16_abt<2><<<dim3(H1 / GEMM_TILE, Bn / GEMM_TILE), 256, 0, stream>>>(
        finalb, fc1wb, fc1b, hidden, Bn, H1, K1, H1);

    // 9) out = hidden @ fc2_w^T + b2 -> f32 [B, 1000]  (B rows padded to 1024)
    gemm_bf16_abt<3><<<dim3(ONp / GEMM_TILE, Bn / GEMM_TILE), 256, 0, stream>>>(
        hidden, fc2wb, fc2b, out, Bn, ON, H1, ON);

    (void)in_sizes; (void)n_in; (void)out_size; (void)ws_size;
}
